// Attention_69801808495122
// MI455X (gfx1250) — matrix operand order repaired and checked by reading, not compiled, not run on hardware
//
#include <hip/hip_runtime.h>

// ---------------------------------------------------------------------------
// Causal multi-head attention forward for MI455X (gfx1250, wave32, WMMA).
//   B=4, N=2048, C=768, H=12, D=64
// Pipeline: qkv GEMM (f32 in -> f16 ws) -> flash attention -> proj GEMM (f32 out)
// Matmuls via v_wmma_f32_16x16x32_f16; pure-f16 tile copies use
// global_load_async_to_lds_b128 (ASYNCcnt) per CDNA5 ISA.
// ---------------------------------------------------------------------------

#define BATCH 4
#define SEQ   2048
#define CDIM  768
#define HEADS 12
#define HDIM  64
#define QKVC  (3 * CDIM)          // 2304
#define MTOT  (BATCH * SEQ)       // 8192
#define EPT   ((size_t)BATCH * HEADS * SEQ * HDIM)  // 6291456 elems per Q/K/V/O

#define LOG2E 1.44269504088896340736f

typedef _Float16 v16h __attribute__((ext_vector_type(16)));
typedef _Float16 v8h  __attribute__((ext_vector_type(8)));
typedef float    v8f  __attribute__((ext_vector_type(8)));

static __device__ __forceinline__ v16h cat8(v8h lo, v8h hi) {
  return __builtin_shufflevector(lo, hi, 0,1,2,3,4,5,6,7,8,9,10,11,12,13,14,15);
}

static __device__ __forceinline__ v8f wmma16(v16h a, v16h b, v8f c) {
  // (neg_a, A, neg_b, B, c_mod, C, reuse_a, reuse_b)
  return __builtin_amdgcn_wmma_f32_16x16x32_f16(false, a, false, b, (short)0, c,
                                                false, false);
}

// Async cache->LDS 16B copy (GVS mode): vdst = LDS byte addr, vaddr = 32-bit
// byte offset, saddr = uniform 64-bit base. Tracked with ASYNCcnt.
static __device__ __forceinline__ void async_copy_b128(const _Float16* lds_dst,
                                                       const void* gbase,
                                                       int voff_bytes) {
  const unsigned lds_addr = (unsigned)(size_t)lds_dst;  // low 32 bits = DS offset
  asm volatile("global_load_async_to_lds_b128 %0, %1, %2"
               :: "v"(lds_addr), "v"(voff_bytes), "s"(gbase)
               : "memory");
}
static __device__ __forceinline__ void wait_async0() {
  asm volatile("s_wait_asynccnt 0" ::: "memory");
}

// A fragment (16x32 f16), LDS row-major [row][k], pitch in elems.
// lane<16: row=lane, elems0..7=k 0..7, elems8..15=k 16..23
// lane>=16: row=lane-16, elems0..7=k 8..15, elems8..15=k 24..31
static __device__ __forceinline__ v16h load_a_frag(const _Float16* base, int pitch,
                                                   int lane, int kOff) {
  const int half = lane >> 4, row = lane & 15;
  const _Float16* p = base + row * pitch + kOff + half * 8;
  return cat8(*(const v8h*)p, *(const v8h*)(p + 16));
}

// B fragment (32x16 f16), LDS stored transposed baseT[col][k], pitch in elems.
// lane<16: col=lane, elems0..7=k 0..7, elems8..15=k 16..23 ; lane>=16: col=lane-16, elems0..7=k 8..15, elems8..15=k 24..31
static __device__ __forceinline__ v16h load_b_frag(const _Float16* baseT, int pitch,
                                                   int lane, int kOff) {
  const int half = lane >> 4, col = lane & 15;
  const _Float16* p = baseT + col * pitch + kOff + half * 8;
  return cat8(*(const v8h*)p, *(const v8h*)(p + 16));
}

// ---------------------------------------------------------------------------
// Kernel 1: QKV = x @ w_qkv ; scatter f16 into Q/K/V [B][H][N][D]
// Block 256 threads (8 waves), tile 128(M) x 128(Ncol), K step 32.
// Wave grid 2(M) x 4(N): each wave 64x32 = 4x2 WMMA tiles.
// grid = (QKVC/128=18, MTOT/128=64)
// ---------------------------------------------------------------------------
__global__ __launch_bounds__(256)
void qkv_gemm_kernel(const float* __restrict__ x, const float* __restrict__ w,
                     _Float16* __restrict__ qg, _Float16* __restrict__ kg,
                     _Float16* __restrict__ vg) {
  __shared__ _Float16 As[128 * 32];
  __shared__ _Float16 Bt[128 * 32];

  const int tid  = threadIdx.x;
  const int lane = tid & 31;
  const int wave = tid >> 5;
  const int wm = wave >> 2;          // 0..1
  const int wn = wave & 3;           // 0..3
  const int m0 = blockIdx.y * 128;
  const int n0 = blockIdx.x * 128;

  v8f acc[4][2];
#pragma unroll
  for (int mi = 0; mi < 4; ++mi)
#pragma unroll
    for (int ni = 0; ni < 2; ++ni)
      acc[mi][ni] = (v8f){0.f,0.f,0.f,0.f,0.f,0.f,0.f,0.f};

  for (int kt = 0; kt < CDIM; kt += 32) {
    // stage A tile 128x32: fp32 -> f16
#pragma unroll
    for (int i = 0; i < 4; ++i) {
      const int r  = (tid >> 3) + i * 32;
      const int kk = (tid & 7) * 4;
      const float4 f = *(const float4*)(x + (size_t)(m0 + r) * CDIM + kt + kk);
      _Float16* dst = As + r * 32 + kk;
      dst[0] = (_Float16)f.x; dst[1] = (_Float16)f.y;
      dst[2] = (_Float16)f.z; dst[3] = (_Float16)f.w;
    }
    // stage B tile 32x128 transposed -> Bt[col][k], fp32 -> f16
#pragma unroll
    for (int i = 0; i < 4; ++i) {
      const int kr = (tid >> 5) + i * 8;
      const int cq = (tid & 31) * 4;
      const float4 f = *(const float4*)(w + (size_t)(kt + kr) * QKVC + n0 + cq);
      Bt[(cq + 0) * 32 + kr] = (_Float16)f.x;
      Bt[(cq + 1) * 32 + kr] = (_Float16)f.y;
      Bt[(cq + 2) * 32 + kr] = (_Float16)f.z;
      Bt[(cq + 3) * 32 + kr] = (_Float16)f.w;
    }
    __syncthreads();

    v16h bfr[2];
#pragma unroll
    for (int ni = 0; ni < 2; ++ni)
      bfr[ni] = load_b_frag(Bt + (wn * 32 + ni * 16) * 32, 32, lane, 0);
#pragma unroll
    for (int mi = 0; mi < 4; ++mi) {
      const v16h afr = load_a_frag(As + (wm * 64 + mi * 16) * 32, 32, lane, 0);
#pragma unroll
      for (int ni = 0; ni < 2; ++ni)
        acc[mi][ni] = wmma16(afr, bfr[ni], acc[mi][ni]);
    }
    __syncthreads();
  }

  // scatter: the 128-col tile lies inside one of q/k/v (768 % 128 == 0), so the
  // segment select and base pointer are block-uniform (SALU, no per-lane div).
  const int s = n0 / CDIM;                        // 0,1,2 (uniform)
  _Float16* __restrict__ dstb = (s == 0) ? qg : (s == 1) ? kg : vg;
  const int colBase = n0 - s * CDIM + wn * 32;    // 0..767 within segment
#pragma unroll
  for (int mi = 0; mi < 4; ++mi)
#pragma unroll
    for (int ni = 0; ni < 2; ++ni)
#pragma unroll
      for (int i = 0; i < 8; ++i) {
        const int mrow = m0 + wm * 64 + mi * 16 + i + 8 * (lane >> 4);
        const int rem  = colBase + ni * 16 + (lane & 15);
        const int h = rem >> 6, d = rem & 63;
        const int b = mrow >> 11, n = mrow & 2047;
        const size_t idx = (((size_t)(b * HEADS + h)) * SEQ + n) * HDIM + d;
        dstb[idx] = (_Float16)acc[mi][ni][i];
      }
}

// ---------------------------------------------------------------------------
// Kernel 2: flash attention. grid = (B*H=48, N/64=32), block 128 (4 waves).
// Each wave owns 16 query rows; streams 64-key tiles with online softmax.
// Q/K tiles staged via async global->LDS copies (ASYNCcnt).
// ---------------------------------------------------------------------------
#define VTP 72  // padded pitch for V-transposed LDS tile

__global__ __launch_bounds__(128)
void flash_attn_kernel(const _Float16* __restrict__ qg,
                       const _Float16* __restrict__ kg,
                       const _Float16* __restrict__ vg,
                       _Float16* __restrict__ og) {
  __shared__ _Float16 Qs[64 * 64];
  __shared__ _Float16 Ks[64 * 64];
  __shared__ _Float16 Vt[64 * VTP];
  __shared__ _Float16 Ps[4 * 16 * 64];

  const int bh = blockIdx.x;
  const int qt = blockIdx.y;
  const int qBase = qt * 64;
  const int tid  = threadIdx.x;
  const int lane = tid & 31;
  const int wave = tid >> 5;

  const _Float16* qp = qg + (size_t)bh * SEQ * HDIM;
  const _Float16* kp = kg + (size_t)bh * SEQ * HDIM;
  const _Float16* vp = vg + (size_t)bh * SEQ * HDIM;

  // async-load Q tile (64x64 f16 = 8KB): 512 x 16B chunks
  {
    const void* qbase = (const void*)(qp + (size_t)qBase * HDIM);
#pragma unroll
    for (int i = 0; i < 4; ++i) {
      const int c = tid + i * 128;
      async_copy_b128(Qs + c * 8, qbase, c * 16);
    }
  }
  wait_async0();
  __syncthreads();

  v16h qf[2];
  qf[0] = load_a_frag(Qs + wave * 16 * 64, 64, lane, 0);
  qf[1] = load_a_frag(Qs + wave * 16 * 64, 64, lane, 32);

  float m[8], l[8];
  v8f accO[4];
#pragma unroll
  for (int i = 0; i < 8; ++i) { m[i] = -3.0e38f; l[i] = 0.f; }
#pragma unroll
  for (int ni = 0; ni < 4; ++ni)
    accO[ni] = (v8f){0.f,0.f,0.f,0.f,0.f,0.f,0.f,0.f};

  _Float16* myP = Ps + wave * 16 * 64;

  for (int jt = 0; jt <= qt; ++jt) {
    const int kBase = jt * 64;

    // async-stage K tile 64x64 flat (natural B layout for QK^T)
    {
      const void* kbase = (const void*)(kp + (size_t)kBase * HDIM);
#pragma unroll
      for (int i = 0; i < 4; ++i) {
        const int c = tid + i * 128;
        async_copy_b128(Ks + c * 8, kbase, c * 16);
      }
    }
    // stage V tile transposed: Vt[d][key] (format shuffle -> manual)
    {
      const int key = tid >> 1, half = tid & 1;
#pragma unroll
      for (int c = 0; c < 4; ++c) {
        const v8h t = *(const v8h*)(vp + (size_t)(kBase + key) * HDIM + half * 32 + c * 8);
#pragma unroll
        for (int e = 0; e < 8; ++e)
          Vt[(half * 32 + c * 8 + e) * VTP + key] = t[e];
      }
    }
    if (jt < qt) {  // prefetch next K/V tiles toward L2
      __builtin_prefetch(kp + (size_t)(kBase + 64) * HDIM + lane * 16, 0, 1);
      __builtin_prefetch(vp + (size_t)(kBase + 64) * HDIM + lane * 16, 0, 1);
    }
    wait_async0();
    __syncthreads();

    // S = Q K^T  (4 key sub-tiles x 2 D-steps)
    v8f sacc[4];
#pragma unroll
    for (int ni = 0; ni < 4; ++ni) {
      const v16h kf0 = load_b_frag(Ks + (ni * 16) * 64, 64, lane, 0);
      const v16h kf1 = load_b_frag(Ks + (ni * 16) * 64, 64, lane, 32);
      v8f sc = (v8f){0.f,0.f,0.f,0.f,0.f,0.f,0.f,0.f};
      sc = wmma16(qf[0], kf0, sc);
      sc = wmma16(qf[1], kf1, sc);
      sacc[ni] = sc;
    }

    // scale + causal mask (only diagonal tile has masked entries)
    const float scale = 0.125f;  // 1/sqrt(64)
#pragma unroll
    for (int ni = 0; ni < 4; ++ni)
#pragma unroll
      for (int i = 0; i < 8; ++i) {
        float sv = sacc[ni][i] * scale;
        if (jt == qt) {
          const int key = kBase + ni * 16 + (lane & 15);
          const int row = qBase + wave * 16 + i + 8 * (lane >> 4);
          if (key > row) sv = -1.0e30f;
        }
        sacc[ni][i] = sv;
      }

    // online softmax: row max across 4 sub-tiles + 16-lane half reduction
    float tmax[8];
#pragma unroll
    for (int i = 0; i < 8; ++i)
      tmax[i] = fmaxf(fmaxf(sacc[0][i], sacc[1][i]), fmaxf(sacc[2][i], sacc[3][i]));
#pragma unroll
    for (int msk = 1; msk < 16; msk <<= 1)
#pragma unroll
      for (int i = 0; i < 8; ++i)
        tmax[i] = fmaxf(tmax[i], __shfl_xor(tmax[i], msk, 32));

    float alpha[8];
#pragma unroll
    for (int i = 0; i < 8; ++i) {
      const float mn = fmaxf(m[i], tmax[i]);
      alpha[i] = exp2f((m[i] - mn) * LOG2E);
      m[i] = mn;
    }

    float rs[8];
#pragma unroll
    for (int i = 0; i < 8; ++i) rs[i] = 0.f;
#pragma unroll
    for (int ni = 0; ni < 4; ++ni)
#pragma unroll
      for (int i = 0; i < 8; ++i) {
        const float p = exp2f((sacc[ni][i] - m[i]) * LOG2E);
        sacc[ni][i] = p;
        rs[i] += p;
      }
#pragma unroll
    for (int msk = 1; msk < 16; msk <<= 1)
#pragma unroll
      for (int i = 0; i < 8; ++i)
        rs[i] += __shfl_xor(rs[i], msk, 32);
#pragma unroll
    for (int i = 0; i < 8; ++i) l[i] = l[i] * alpha[i] + rs[i];
#pragma unroll
    for (int ni = 0; ni < 4; ++ni)
#pragma unroll
      for (int i = 0; i < 8; ++i) accO[ni][i] *= alpha[i];

    // spill P to per-wave LDS to re-swizzle C/D layout -> A-fragment layout
#pragma unroll
    for (int ni = 0; ni < 4; ++ni)
#pragma unroll
      for (int i = 0; i < 8; ++i) {
        const int row = i + 8 * (lane >> 4);
        const int colp = ni * 16 + (lane & 15);
        myP[row * 64 + colp] = (_Float16)sacc[ni][i];
      }

    // O += P @ V  (2 key-chunks x 4 D sub-tiles)
#pragma unroll
    for (int c = 0; c < 2; ++c) {
      const v16h pf = load_a_frag(myP, 64, lane, c * 32);
#pragma unroll
      for (int ni = 0; ni < 4; ++ni) {
        const v16h vf = load_b_frag(Vt + (ni * 16) * VTP, VTP, lane, c * 32);
        accO[ni] = wmma16(pf, vf, accO[ni]);
      }
    }
    __syncthreads();
  }

  // epilogue: O /= l, write f16 O as [B][N][H][D]
  float invl[8];
#pragma unroll
  for (int i = 0; i < 8; ++i) invl[i] = 1.0f / l[i];
  const int b = bh / HEADS, h = bh % HEADS;
#pragma unroll
  for (int ni = 0; ni < 4; ++ni)
#pragma unroll
    for (int i = 0; i < 8; ++i) {
      const int row = qBase + wave * 16 + i + 8 * (lane >> 4);
      const int d = ni * 16 + (lane & 15);
      og[(((size_t)b * SEQ + row) * HEADS + h) * HDIM + d] =
          (_Float16)(accO[ni][i] * invl[i]);
    }
}

// ---------------------------------------------------------------------------
// Kernel 3: out = O @ w_proj + b_proj (fp32 out). grid = (CDIM/128=6, 64).
// A tile (already f16) staged via async global->LDS copies.
// ---------------------------------------------------------------------------
__global__ __launch_bounds__(256)
void proj_gemm_kernel(const _Float16* __restrict__ og, const float* __restrict__ w,
                      const float* __restrict__ bias, float* __restrict__ out) {
  __shared__ _Float16 As[128 * 32];
  __shared__ _Float16 Bt[128 * 32];

  const int tid  = threadIdx.x;
  const int lane = tid & 31;
  const int wave = tid >> 5;
  const int wm = wave >> 2;
  const int wn = wave & 3;
  const int m0 = blockIdx.y * 128;
  const int n0 = blockIdx.x * 128;

  v8f acc[4][2];
#pragma unroll
  for (int mi = 0; mi < 4; ++mi)
#pragma unroll
    for (int ni = 0; ni < 2; ++ni)
      acc[mi][ni] = (v8f){0.f,0.f,0.f,0.f,0.f,0.f,0.f,0.f};

  for (int kt = 0; kt < CDIM; kt += 32) {
    // A tile 128x32 f16: async copy (strided rows, 16B chunks)
    {
      const void* abase = (const void*)(og + (size_t)m0 * CDIM + kt);
#pragma unroll
      for (int i = 0; i < 2; ++i) {
        const int idx16 = tid + i * 256;         // 512 chunks of 8 f16
        const int r  = idx16 >> 2;
        const int kk = (idx16 & 3) * 8;
        async_copy_b128(As + r * 32 + kk, abase, (r * CDIM + kk) * 2);
      }
    }
    // B tile 32x128 transposed, fp32 -> f16
#pragma unroll
    for (int i = 0; i < 4; ++i) {
      const int kr = (tid >> 5) + i * 8;
      const int cq = (tid & 31) * 4;
      const float4 f = *(const float4*)(w + (size_t)(kt + kr) * CDIM + n0 + cq);
      Bt[(cq + 0) * 32 + kr] = (_Float16)f.x;
      Bt[(cq + 1) * 32 + kr] = (_Float16)f.y;
      Bt[(cq + 2) * 32 + kr] = (_Float16)f.z;
      Bt[(cq + 3) * 32 + kr] = (_Float16)f.w;
    }
    wait_async0();
    __syncthreads();

    v16h bfr[2];
#pragma unroll
    for (int ni = 0; ni < 2; ++ni)
      bfr[ni] = load_b_frag(Bt + (wn * 32 + ni * 16) * 32, 32, lane, 0);
#pragma unroll
    for (int mi = 0; mi < 4; ++mi) {
      const v16h afr = load_a_frag(As + (wm * 64 + mi * 16) * 32, 32, lane, 0);
#pragma unroll
      for (int ni = 0; ni < 2; ++ni)
        acc[mi][ni] = wmma16(afr, bfr[ni], acc[mi][ni]);
    }
    __syncthreads();
  }

#pragma unroll
  for (int mi = 0; mi < 4; ++mi)
#pragma unroll
    for (int ni = 0; ni < 2; ++ni)
#pragma unroll
      for (int i = 0; i < 8; ++i) {
        const int mrow = m0 + wm * 64 + mi * 16 + i + 8 * (lane >> 4);
        const int col  = n0 + wn * 32 + ni * 16 + (lane & 15);
        out[(size_t)mrow * CDIM + col] = acc[mi][ni][i] + bias[col];
      }
}

// ---------------------------------------------------------------------------
extern "C" void kernel_launch(void* const* d_in, const int* in_sizes, int n_in,
                              void* d_out, int out_size, void* d_ws, size_t ws_size,
                              hipStream_t stream) {
  const float* x      = (const float*)d_in[0];
  const float* w_qkv  = (const float*)d_in[1];
  const float* w_proj = (const float*)d_in[2];
  const float* b_proj = (const float*)d_in[3];
  float* out = (float*)d_out;

  // workspace: Q,K,V,O each EPT f16 elems (total ~50.3 MB)
  _Float16* ws = (_Float16*)d_ws;
  _Float16* Q = ws;
  _Float16* K = ws + EPT;
  _Float16* V = ws + 2 * EPT;
  _Float16* O = ws + 3 * EPT;

  qkv_gemm_kernel<<<dim3(QKVC / 128, MTOT / 128), 256, 0, stream>>>(x, w_qkv, Q, K, V);
  flash_attn_kernel<<<dim3(BATCH * HEADS, SEQ / 64), 128, 0, stream>>>(Q, K, V, O);
  proj_gemm_kernel<<<dim3(CDIM / 128, MTOT / 128), 256, 0, stream>>>(O, w_proj, b_proj, out);
}
